// SelfAttention_12695923327378
// MI455X (gfx1250) — matrix operand order repaired and checked by reading, not compiled, not run on hardware
//
#include <hip/hip_runtime.h>

// ---------------------------------------------------------------------------
// Problem constants (match reference)
// ---------------------------------------------------------------------------
#define SEQ_LEN 4096
#define D_EMBED 1024
#define N_HEADS 16
#define D_HEAD  64
#define TRIPLE_D (3 * D_EMBED)

typedef __bf16 bf16;
typedef __attribute__((ext_vector_type(16))) __bf16 v16bf;
typedef __attribute__((ext_vector_type(8)))  __bf16 v8bf;
typedef __attribute__((ext_vector_type(8)))  float  v8f;

// ---------------------------------------------------------------------------
// WMMA helper: D = A(16x32 bf16) * B(32x16 bf16) + C(16x16 f32)
// ---------------------------------------------------------------------------
static __device__ inline v8f wmma_bf16(v16bf a, v16bf b, v8f c) {
    return __builtin_amdgcn_wmma_f32_16x16x32_bf16(
        /*neg_a=*/false, a, /*neg_b=*/false, b,
        /*c_mod=*/(short)0, c, /*reuse_a=*/false, /*reuse_b=*/false);
}

static __device__ inline v16bf combine16(v8bf lo, v8bf hi) {
    v16bf r;
#pragma unroll
    for (int i = 0; i < 8; ++i) { r[i] = lo[i]; r[i + 8] = hi[i]; }
    return r;
}

// A-matrix fragment (16x32 bf16, MxK). Memory: row-major [M][K], row stride
// `stride` elements, base -> element (0,0) of the tile.
// Layout (ISA 7.12.2): lanes 0-15 hold M=lane, K in {0..7, 16..23};
//                      lanes 16-31 hold M=lane-16, K in {8..15, 24..31}.
static __device__ inline v16bf load_a_frag(const bf16* base, int stride, int lane) {
    const int m  = lane & 15;
    const int kb = (lane >> 4) ? 8 : 0;
    v8bf lo = *(const v8bf*)(base + (size_t)m * stride + kb);
    v8bf hi = *(const v8bf*)(base + (size_t)m * stride + kb + 16);
    return combine16(lo, hi);
}

// B-matrix fragment (32x16 bf16, KxN). Memory: element (K,N) at
// base[N*stride + K] (i.e. K-contiguous / "B-transposed" storage).
// Layout: lanes 0-15 hold N=lane, K in {0..7, 16..23}; lanes 16-31 hold N=lane-16, K in {8..15, 24..31}.
static __device__ inline v16bf load_b_frag(const bf16* base, int stride, int lane) {
    const int n  = lane & 15;
    const int kb = (lane >> 4) ? 8 : 0;
    v8bf lo = *(const v8bf*)(base + (size_t)n * stride + kb);
    v8bf hi = *(const v8bf*)(base + (size_t)n * stride + kb + 16);
    return combine16(lo, hi);
}

// Cross-lane reductions over the 16-lane half-wave (matches C-layout N dim).
static __device__ inline float rowmax16(float v) {
#pragma unroll
    for (int m = 8; m >= 1; m >>= 1) v = fmaxf(v, __shfl_xor(v, m, 16));
    return v;
}
static __device__ inline float rowsum16(float v) {
#pragma unroll
    for (int m = 8; m >= 1; m >>= 1) v += __shfl_xor(v, m, 16);
    return v;
}

// Async 16B copy global -> LDS (ASYNCcnt-tracked; gfx1250).
// ldsOff = byte offset in LDS (low 32 bits of a generic shared pointer).
static __device__ inline void async_copy_b128(unsigned int ldsOff,
                                              unsigned long long gaddr) {
    asm volatile("global_load_async_to_lds_b128 %0, %1, off"
                 :: "v"(ldsOff), "v"(gaddr)
                 : "memory");
}
static __device__ inline void wait_asynccnt0() {
    asm volatile("s_wait_asynccnt 0x0" ::: "memory");
}

// ---------------------------------------------------------------------------
// f32 -> bf16 conversion kernels
// ---------------------------------------------------------------------------
__global__ void attn_cvt_bf16(const float* __restrict__ src,
                              bf16* __restrict__ dst, int n) {
    int stride = gridDim.x * blockDim.x;
    for (int i = blockIdx.x * blockDim.x + threadIdx.x; i < n; i += stride)
        dst[i] = (bf16)src[i];
}

// src[rows][cols] (f32) -> dst[cols][rows] (bf16)
__global__ void attn_transpose_cvt_bf16(const float* __restrict__ src,
                                        bf16* __restrict__ dst,
                                        int rows, int cols) {
    int stride = gridDim.x * blockDim.x;
    int total = rows * cols;
    for (int i = blockIdx.x * blockDim.x + threadIdx.x; i < total; i += stride) {
        int r = i / cols;
        int c = i - r * cols;
        dst[(size_t)c * rows + r] = (bf16)src[i];
    }
}

// ---------------------------------------------------------------------------
// WMMA GEMM: C[M][N] = A[M][K] @ B[K][N] + bias[N]
//   A  : bf16 row-major [M][K]
//   Bt : bf16 [N][K]  (B transposed -> K-contiguous B fragments)
//   C  : bf16 or f32, row-major [M][N]
// Block = 128 threads (4 waves); each wave computes a 16x64 strip.
// ---------------------------------------------------------------------------
template <bool STORE_BF16>
__global__ __launch_bounds__(128)
void attn_wmma_gemm(const bf16* __restrict__ A,
                    const bf16* __restrict__ Bt,
                    const float* __restrict__ bias,
                    void* __restrict__ Cout,
                    int M, int N, int K) {
    const int lane = threadIdx.x & 31;
    const int wave = threadIdx.x >> 5;
    const int n0 = blockIdx.x * 64;
    const int m0 = (blockIdx.y * 4 + wave) * 16;

    v8f acc[4] = {};
    for (int k0 = 0; k0 < K; k0 += 32) {
        v16bf af = load_a_frag(A + (size_t)m0 * K + k0, K, lane);
#pragma unroll
        for (int c = 0; c < 4; ++c) {
            v16bf bfg = load_b_frag(Bt + (size_t)(n0 + c * 16) * K + k0, K, lane);
            acc[c] = wmma_bf16(af, bfg, acc[c]);
        }
    }

    const int rowbase = m0 + 8 * (lane >> 4);
    const int colbase = n0 + (lane & 15);
#pragma unroll
    for (int c = 0; c < 4; ++c) {
        const int col = colbase + c * 16;
        const float bv = bias[col];
#pragma unroll
        for (int r = 0; r < 8; ++r) {
            float v = acc[c][r] + bv;
            if (STORE_BF16)
                ((bf16*)Cout)[(size_t)(rowbase + r) * N + col] = (bf16)v;
            else
                ((float*)Cout)[(size_t)(rowbase + r) * N + col] = v;
        }
    }
}

// ---------------------------------------------------------------------------
// Split qkv [N][3D] bf16 into per-head q [H][N][dh], k [H][N][dh],
// and v transposed [H][dh][N] (so P@V B-fragments are K-contiguous).
// ---------------------------------------------------------------------------
__global__ void attn_split_qkv(const bf16* __restrict__ qkv,
                               bf16* __restrict__ qh,
                               bf16* __restrict__ kh,
                               bf16* __restrict__ vt) {
    const int total = N_HEADS * SEQ_LEN * D_HEAD;
    int i = blockIdx.x * blockDim.x + threadIdx.x;
    if (i >= total) return;
    int h = i / (SEQ_LEN * D_HEAD);
    int rem = i - h * SEQ_LEN * D_HEAD;
    int n = rem / D_HEAD;
    int d = rem - n * D_HEAD;
    size_t src = (size_t)n * TRIPLE_D + h * D_HEAD + d;
    bf16 qv = qkv[src];
    bf16 kv = qkv[src + D_EMBED];
    bf16 vv = qkv[src + 2 * D_EMBED];
    qh[(size_t)h * SEQ_LEN * D_HEAD + (size_t)n * D_HEAD + d] = qv;
    kh[(size_t)h * SEQ_LEN * D_HEAD + (size_t)n * D_HEAD + d] = kv;
    vt[(size_t)h * D_HEAD * SEQ_LEN + (size_t)d * SEQ_LEN + n] = vv;
}

// ---------------------------------------------------------------------------
// Flash-style masked attention. Grid: (H, SEQ_LEN/64), block 128 (4 waves).
// All 4 waves of a block share the same head and the same K/V stream, so the
// K tile (32x64 bf16, 4KB) and V^T tile (64x32 bf16, 4KB) are staged once per
// step into LDS with GLOBAL_LOAD_ASYNC_TO_LDS_B128 (ASYNCcnt), then all
// fragment reads hit LDS (ds_load_b128) -> 4x less L2 traffic.
// P tile (C layout) is re-laid-out to A layout through a per-wave LDS buffer.
// ---------------------------------------------------------------------------
#define MASK_NEG (-1.0e30f)
#define PROW_STRIDE 40   // 16-byte-aligned LDS row stride (bf16 elements)

__global__ __launch_bounds__(128)
void attn_flash(const bf16* __restrict__ qh,
                const bf16* __restrict__ kh,
                const bf16* __restrict__ vt,
                const int*  __restrict__ batch,
                bf16* __restrict__ aout) {
    const int tid  = threadIdx.x;
    const int lane = tid & 31;
    const int wave = tid >> 5;
    const int h  = blockIdx.x;
    const int q0 = (blockIdx.y * 4 + wave) * 16;
    const int halfup = lane >> 4;
    const int nlo = lane & 15;

    const bf16* qhh = qh + (size_t)h * SEQ_LEN * D_HEAD;
    const bf16* khh = kh + (size_t)h * SEQ_LEN * D_HEAD;
    const bf16* vth = vt + (size_t)h * D_HEAD * SEQ_LEN;

    __shared__ __attribute__((aligned(16))) bf16 ldsK[32 * D_HEAD];   // [m][d]
    __shared__ __attribute__((aligned(16))) bf16 ldsV[D_HEAD * 32];   // [d][m]
    __shared__ __attribute__((aligned(16))) bf16 pbuf[4][16 * PROW_STRIDE];
    bf16* pw = pbuf[wave];

    // Per-thread async-copy plan: 512 16B chunks per step (256 K + 256 V),
    // 4 chunks per thread. LDS offsets fixed; global pointers advance per step.
    const bf16*        gptr[4];
    long               gstep[4];
    unsigned int       loff[4];
#pragma unroll
    for (int j = 0; j < 4; ++j) {
        const int c = tid + j * 128;
        if (c < 256) {                       // K tile: row m (0..31), 8 chunks/row
            const int row = c >> 3, col = c & 7;
            gptr[j]  = khh + (size_t)row * D_HEAD + col * 8;
            gstep[j] = 32 * D_HEAD;          // m0 += 32
            loff[j]  = (unsigned int)(unsigned long long)(&ldsK[c * 8]);
        } else {                             // V tile: row d (0..63), 4 chunks/row
            const int cc = c - 256;
            const int row = cc >> 2, qq = cc & 3;
            gptr[j]  = vth + (size_t)row * SEQ_LEN + qq * 8;
            gstep[j] = 32;                   // m0 += 32
            loff[j]  = (unsigned int)(unsigned long long)(&ldsV[cc * 8]);
        }
    }

    // Q fragments for K = d 0..31 and 32..63 (reused across the whole loop).
    const v16bf aq0 = load_a_frag(qhh + (size_t)q0 * D_HEAD, D_HEAD, lane);
    const v16bf aq1 = load_a_frag(qhh + (size_t)q0 * D_HEAD + 32, D_HEAD, lane);

    float mrow[8], lrow[8];
    int brow[8];
    v8f acc[4] = {};
#pragma unroll
    for (int r = 0; r < 8; ++r) {
        mrow[r] = MASK_NEG;
        lrow[r] = 0.0f;
        brow[r] = batch[q0 + r + 8 * halfup];
    }

    for (int m0 = 0; m0 < SEQ_LEN; m0 += 32) {
        // ---- cooperative async stage of K / V^T tiles into LDS ----
#pragma unroll
        for (int j = 0; j < 4; ++j) {
            async_copy_b128(loff[j], (unsigned long long)gptr[j]);
            gptr[j] += gstep[j];
        }
        wait_asynccnt0();
        __syncthreads();

        // S tiles: s0 covers key cols m0..m0+15, s1 covers m0+16..m0+31.
        v8f s0 = {}, s1 = {};
        s0 = wmma_bf16(aq0, load_b_frag(ldsK, D_HEAD, lane), s0);
        s0 = wmma_bf16(aq1, load_b_frag(ldsK + 32, D_HEAD, lane), s0);
        s1 = wmma_bf16(aq0, load_b_frag(ldsK + 16 * D_HEAD, D_HEAD, lane), s1);
        s1 = wmma_bf16(aq1, load_b_frag(ldsK + 16 * D_HEAD + 32, D_HEAD, lane), s1);

        const int bc0 = batch[m0 + nlo];
        const int bc1 = batch[m0 + 16 + nlo];

        float p0[8], p1[8];
#pragma unroll
        for (int r = 0; r < 8; ++r) {
            float v0 = (brow[r] == bc0) ? (s0[r] + 1.0f) * 0.125f : MASK_NEG;
            float v1 = (brow[r] == bc1) ? (s1[r] + 1.0f) * 0.125f : MASK_NEG;
            float mx = rowmax16(fmaxf(v0, v1));
            float mnew = fmaxf(mrow[r], mx);
            float alpha = __expf(mrow[r] - mnew);
            float e0 = __expf(v0 - mnew);
            float e1 = __expf(v1 - mnew);
            float rs = rowsum16(e0 + e1);
            lrow[r] = lrow[r] * alpha + rs;
            mrow[r] = mnew;
            p0[r] = e0;
            p1[r] = e1;
#pragma unroll
            for (int c = 0; c < 4; ++c) acc[c][r] *= alpha;
        }

        // C layout -> A layout for P via per-wave LDS staging (same-wave DS
        // ops are kept in order by hardware).
#pragma unroll
        for (int r = 0; r < 8; ++r) {
            const int row = r + 8 * halfup;
            pw[row * PROW_STRIDE + nlo]      = (bf16)p0[r];
            pw[row * PROW_STRIDE + 16 + nlo] = (bf16)p1[r];
        }
        v16bf pa = load_a_frag(pw, PROW_STRIDE, lane);

        // O += P(16x32) @ V(32x64): V^T tile in LDS -> K-contiguous B frags.
#pragma unroll
        for (int c = 0; c < 4; ++c) {
            v16bf vf = load_b_frag(ldsV + (c * 16) * 32, 32, lane);
            acc[c] = wmma_bf16(pa, vf, acc[c]);
        }

        __syncthreads();   // protect LDS tiles before next stage overwrites
    }

    // Epilogue: normalize and store into per-token [N][D] bf16 layout.
#pragma unroll
    for (int r = 0; r < 8; ++r) {
        const float inv = 1.0f / lrow[r];
        const int row = q0 + r + 8 * halfup;
#pragma unroll
        for (int c = 0; c < 4; ++c)
            aout[(size_t)row * D_EMBED + h * D_HEAD + c * 16 + nlo] =
                (bf16)(acc[c][r] * inv);
    }
}

// ---------------------------------------------------------------------------
// Host launcher
// ---------------------------------------------------------------------------
extern "C" void kernel_launch(void* const* d_in, const int* in_sizes, int n_in,
                              void* d_out, int out_size, void* d_ws, size_t ws_size,
                              hipStream_t stream) {
    const float* x     = (const float*)d_in[0];
    const int*   batch = (const int*)d_in[1];
    const float* W_in  = (const float*)d_in[2];
    const float* b_in  = (const float*)d_in[3];
    const float* W_out = (const float*)d_in[4];
    const float* b_out = (const float*)d_in[5];
    float* out = (float*)d_out;

    // Workspace carve-up (256B aligned slices).
    char* ws = (char*)d_ws;
    size_t off = 0;
    auto alloc = [&](size_t bytes) -> void* {
        void* p = ws + off;
        off += (bytes + 255) & ~(size_t)255;
        return p;
    };
    bf16* xb    = (bf16*)alloc((size_t)SEQ_LEN * D_EMBED * 2);     //  8 MB
    bf16* WtIn  = (bf16*)alloc((size_t)TRIPLE_D * D_EMBED * 2);    //  6 MB
    bf16* WtOut = (bf16*)alloc((size_t)D_EMBED * D_EMBED * 2);     //  2 MB
    bf16* qkvb  = (bf16*)alloc((size_t)SEQ_LEN * TRIPLE_D * 2);    // 24 MB
    bf16* qhb   = (bf16*)alloc((size_t)N_HEADS * SEQ_LEN * D_HEAD * 2); // 8 MB
    bf16* khb   = (bf16*)alloc((size_t)N_HEADS * SEQ_LEN * D_HEAD * 2); // 8 MB
    bf16* vtb   = (bf16*)alloc((size_t)N_HEADS * D_HEAD * SEQ_LEN * 2); // 8 MB
    bf16* aoutb = (bf16*)alloc((size_t)SEQ_LEN * D_EMBED * 2);     //  8 MB

    // 1) Convert activations / (transposed) weights to bf16.
    attn_cvt_bf16<<<2048, 256, 0, stream>>>(x, xb, SEQ_LEN * D_EMBED);
    attn_transpose_cvt_bf16<<<2048, 256, 0, stream>>>(W_in, WtIn, D_EMBED, TRIPLE_D);
    attn_transpose_cvt_bf16<<<1024, 256, 0, stream>>>(W_out, WtOut, D_EMBED, D_EMBED);

    // 2) QKV projection: qkv = x @ W_in + b_in  (bf16 out).
    attn_wmma_gemm<true><<<dim3(TRIPLE_D / 64, SEQ_LEN / 64), 128, 0, stream>>>(
        xb, WtIn, b_in, qkvb, SEQ_LEN, TRIPLE_D, D_EMBED);

    // 3) Split into per-head q / k / v-transposed.
    attn_split_qkv<<<(N_HEADS * SEQ_LEN * D_HEAD) / 256, 256, 0, stream>>>(
        qkvb, qhb, khb, vtb);

    // 4) Masked flash attention per head (async-LDS staged K/V).
    attn_flash<<<dim3(N_HEADS, SEQ_LEN / 64), 128, 0, stream>>>(
        qhb, khb, vtb, batch, aoutb);

    // 5) Output projection: out = attn @ W_out + b_out  (f32 out).
    attn_wmma_gemm<false><<<dim3(D_EMBED / 64, SEQ_LEN / 64), 128, 0, stream>>>(
        aoutb, WtOut, b_out, out, SEQ_LEN, D_EMBED, D_EMBED);
}
